// RBF_11081015623693
// MI455X (gfx1250) — matrix operand order repaired and checked by reading, not compiled, not run on hardware
//
#include <hip/hip_runtime.h>
#include <hip/hip_bf16.h>

// Problem sizes (match reference)
#define B_DIM 4096
#define C_DIM 4096
#define D_DIM 1024
#define COLB  8                        // column-slice blocks over C
#define CSLICE (C_DIM / COLB)          // 512 centers per (blockIdx.y) slice

typedef __attribute__((ext_vector_type(16))) __bf16 v16bf;
typedef __attribute__((ext_vector_type(8)))  __bf16 v8bf;
typedef __attribute__((ext_vector_type(4)))  __bf16 v4bf;
typedef __attribute__((ext_vector_type(8)))  float  v8f;

#define WMMA_BF16(A, Bf, Cacc) \
    __builtin_amdgcn_wmma_f32_16x16x32_bf16(false, (A), false, (Bf), (short)0, (Cacc), false, false)

// A-fragment: elements 0..15 at p[0..15]
__device__ __forceinline__ v16bf load_a16(const __bf16* __restrict__ p) {
    v8bf a = *(const v8bf*)p;
    v8bf b = *(const v8bf*)(p + 8);
    return __builtin_shufflevector(a, b, 0,1,2,3,4,5,6,7,8,9,10,11,12,13,14,15);
}
// B-fragment: 16 contiguous bf16
__device__ __forceinline__ v16bf load_b16(const __bf16* __restrict__ p) {
    return *(const v16bf*)p;
}

// ---------------------------------------------------------------------------
// Kernel 0: split f32 tensors into bf16 hi/lo planes (done ONCE, outside GEMM).
// First NX4 float4s belong to x, rest to centers.
// ---------------------------------------------------------------------------
__global__ __launch_bounds__(256)
void cvt_split_kernel(const float* __restrict__ x, const float* __restrict__ cen,
                      __bf16* __restrict__ xhi, __bf16* __restrict__ xlo,
                      __bf16* __restrict__ chi, __bf16* __restrict__ clo) {
    const int NX4 = B_DIM * D_DIM / 4;   // float4 count for x
    int idx = blockIdx.x * 256 + threadIdx.x;
    const float* src;
    __bf16 *dhi, *dlo;
    int off;
    if (idx < NX4) { src = x;  dhi = xhi; dlo = xlo; off = idx; }
    else           { src = cen; dhi = chi; dlo = clo; off = idx - NX4; }
    float4 v = reinterpret_cast<const float4*>(src)[off];
    float e[4] = { v.x, v.y, v.z, v.w };
    v4bf h4, l4;
#pragma unroll
    for (int j = 0; j < 4; ++j) {
        __bf16 h = (__bf16)e[j];               // RTNE high part
        __bf16 l = (__bf16)(e[j] - (float)h);  // residual low part
        h4[j] = h; l4[j] = l;
    }
    *reinterpret_cast<v4bf*>(dhi + (size_t)off * 4) = h4;
    *reinterpret_cast<v4bf*>(dlo + (size_t)off * 4) = l4;
}

// ---------------------------------------------------------------------------
// Kernel 1: row squared-norms for x (B rows) and centers (C rows).
// ---------------------------------------------------------------------------
__global__ __launch_bounds__(256)
void rownorm_kernel(const float* __restrict__ x, const float* __restrict__ cen,
                    float* __restrict__ x2, float* __restrict__ c2) {
    int wave = threadIdx.x >> 5;
    int lane = threadIdx.x & 31;
    int row  = blockIdx.x * 8 + wave;
    const float* p = (row < B_DIM) ? (x + (size_t)row * D_DIM)
                                   : (cen + (size_t)(row - B_DIM) * D_DIM);
    float s = 0.f;
#pragma unroll
    for (int i = 0; i < 8; ++i) {
        float4 q = reinterpret_cast<const float4*>(p)[i * 32 + lane];
        s += q.x * q.x + q.y * q.y + q.z * q.z + q.w * q.w;
    }
#pragma unroll
    for (int off = 1; off < 32; off <<= 1) s += __shfl_xor(s, off, 32);
    if (lane == 0) {
        if (row < B_DIM) x2[row] = s;
        else             c2[row - B_DIM] = s;
    }
}

// ---------------------------------------------------------------------------
// Kernel 2: fused GEMM (split-bf16 WMMA from preconverted planes) + epilogue.
// Wave tile: 32 rows x 64 cols (2x4 16x16 tiles, 3 WMMAs each per K-step = 24).
// Workgroup: 8 waves = 2 (rows) x 4 (cols) -> 64 rows x 256 cols per iteration.
// blockIdx.x: row block (64 rows); blockIdx.y: column slice of C (512 centers).
// ---------------------------------------------------------------------------
__global__ __launch_bounds__(256)
void rbf_partial_kernel(const __bf16* __restrict__ xhi, const __bf16* __restrict__ xlo,
                        const __bf16* __restrict__ chi, const __bf16* __restrict__ clo,
                        const float* __restrict__ beta, const float* __restrict__ W,
                        const float* __restrict__ x2,  const float* __restrict__ c2,
                        float* __restrict__ partial) {
    __shared__ float red[4][64];

    const int tid  = threadIdx.x;
    const int wave = tid >> 5;
    const int lane = tid & 31;
    const int wr   = wave >> 2;        // 0..1 row position
    const int wc   = wave & 3;         // 0..3 col position
    const int laneRow = lane & 15;
    const int halfsel = lane >> 4;     // 0 or 1

    const int rowBase = blockIdx.x * 64 + wr * 32;
    const int colSliceBase = blockIdx.y * CSLICE;

    // x2 for the 16 rows this lane's accumulator elements map to.
    float ax2[2][8];
#pragma unroll
    for (int i = 0; i < 2; ++i)
#pragma unroll
        for (int v = 0; v < 8; ++v)
            ax2[i][v] = x2[rowBase + i * 16 + halfsel * 8 + v];

    float psum[2][8];
#pragma unroll
    for (int i = 0; i < 2; ++i)
#pragma unroll
        for (int v = 0; v < 8; ++v) psum[i][v] = 0.f;

    // A-lane pointers: row = rowBase + i*16 + laneRow, K-offset 16*halfsel.
    const size_t arow = (size_t)(rowBase + laneRow) * D_DIM + halfsel * 16;
    const __bf16* pa_hi0 = xhi + arow;
    const __bf16* pa_lo0 = xlo + arow;
    const __bf16* pa_hi1 = pa_hi0 + (size_t)16 * D_DIM;
    const __bf16* pa_lo1 = pa_lo0 + (size_t)16 * D_DIM;

    for (int ci = 0; ci < CSLICE / 256; ++ci) {
        const int colBase = colSliceBase + ci * 256 + wc * 64;

        const __bf16* pb_hi[4];
        const __bf16* pb_lo[4];
#pragma unroll
        for (int j = 0; j < 4; ++j) {
            const size_t brow = (size_t)(colBase + j * 16 + laneRow) * D_DIM + halfsel * 16;
            pb_hi[j] = chi + brow;
            pb_lo[j] = clo + brow;
        }

        v8f acc[2][4];
#pragma unroll
        for (int i = 0; i < 2; ++i)
#pragma unroll
            for (int j = 0; j < 4; ++j) acc[i][j] = (v8f){};

        for (int kb = 0; kb < D_DIM; kb += 32) {
            v16bf fa_hi0 = load_a16(pa_hi0 + kb);
            v16bf fa_lo0 = load_a16(pa_lo0 + kb);
            v16bf fa_hi1 = load_a16(pa_hi1 + kb);
            v16bf fa_lo1 = load_a16(pa_lo1 + kb);
#pragma unroll
            for (int j = 0; j < 4; ++j) {
                v16bf fb_hi = load_b16(pb_hi[j] + kb);
                v16bf fb_lo = load_b16(pb_lo[j] + kb);
                // hi*hi + hi*lo + lo*hi (~f32 accuracy, f32 accumulate)
                acc[0][j] = WMMA_BF16(fa_hi0, fb_hi, acc[0][j]);
                acc[0][j] = WMMA_BF16(fa_hi0, fb_lo, acc[0][j]);
                acc[0][j] = WMMA_BF16(fa_lo0, fb_hi, acc[0][j]);
                acc[1][j] = WMMA_BF16(fa_hi1, fb_hi, acc[1][j]);
                acc[1][j] = WMMA_BF16(fa_hi1, fb_lo, acc[1][j]);
                acc[1][j] = WMMA_BF16(fa_lo1, fb_hi, acc[1][j]);
            }
        }

        // Epilogue: d2 = x2 + c2 - 2*dot; h = exp(-beta*sqrt(max(d2,0))); psum += h*W[c]
#pragma unroll
        for (int j = 0; j < 4; ++j) {
            const int col = colBase + j * 16 + laneRow;
            const float bv = beta[col];
            const float wv = W[col];
            const float cc = c2[col];
#pragma unroll
            for (int i = 0; i < 2; ++i) {
#pragma unroll
                for (int v = 0; v < 8; ++v) {
                    float d2v = fmaxf(ax2[i][v] + cc - 2.0f * acc[i][j][v], 0.f);
                    float h = __expf(-bv * __builtin_sqrtf(d2v));
                    if (__builtin_isinf(h)) h = 0.f;
                    psum[i][v] += h * wv;
                }
            }
        }
    }

    // Reduce psum across the 16-lane half-groups (columns within the wave).
#pragma unroll
    for (int off = 1; off < 16; off <<= 1) {
#pragma unroll
        for (int i = 0; i < 2; ++i)
#pragma unroll
            for (int v = 0; v < 8; ++v)
                psum[i][v] += __shfl_xor(psum[i][v], off, 32);
    }

    // Lane 0 holds rows (base..base+7), lane 16 holds rows (base+8..base+15).
    if (laneRow == 0) {
#pragma unroll
        for (int i = 0; i < 2; ++i)
#pragma unroll
            for (int v = 0; v < 8; ++v)
                red[wc][wr * 32 + i * 16 + halfsel * 8 + v] = psum[i][v];
    }
    __syncthreads();

    // Combine the 4 column-waves in a fixed order (deterministic).
    if (tid < 64) {
        float s = red[0][tid] + red[1][tid] + red[2][tid] + red[3][tid];
        partial[(size_t)blockIdx.y * B_DIM + blockIdx.x * 64 + tid] = s;
    }
}

// ---------------------------------------------------------------------------
// Kernel 3: reduce the COLB column-slice partials per row, add bias.
// ---------------------------------------------------------------------------
__global__ __launch_bounds__(256)
void rbf_final_kernel(const float* __restrict__ partial, const float* __restrict__ bias,
                      float* __restrict__ out) {
    int idx = blockIdx.x * 256 + threadIdx.x;
    if (idx < B_DIM) {
        float s = bias[0];
#pragma unroll
        for (int cb = 0; cb < COLB; ++cb) s += partial[(size_t)cb * B_DIM + idx];
        out[idx] = s;
    }
}

extern "C" void kernel_launch(void* const* d_in, const int* in_sizes, int n_in,
                              void* d_out, int out_size, void* d_ws, size_t ws_size,
                              hipStream_t stream) {
    const float* x    = (const float*)d_in[0];  // [B, D]
    const float* cen  = (const float*)d_in[1];  // [C, D]
    const float* beta = (const float*)d_in[2];  // [C]
    const float* W    = (const float*)d_in[3];  // [1, C]
    const float* b    = (const float*)d_in[4];  // [1]
    float* out = (float*)d_out;                 // [B, 1]

    // Workspace layout:
    //   f32: x2[B], c2[C], partial[COLB*B]            -> 40960 floats (160 KB)
    //   bf16: xhi[B*D], xlo[B*D], chi[C*D], clo[C*D]  -> 32 MB
    float* ws      = (float*)d_ws;
    float* x2      = ws;
    float* c2      = ws + B_DIM;
    float* partial = ws + B_DIM + C_DIM;
    __bf16* bfbase = (__bf16*)(ws + B_DIM + C_DIM + (size_t)COLB * B_DIM);
    __bf16* xhi = bfbase;
    __bf16* xlo = xhi + (size_t)B_DIM * D_DIM;
    __bf16* chi = xlo + (size_t)B_DIM * D_DIM;
    __bf16* clo = chi + (size_t)C_DIM * D_DIM;

    (void)in_sizes; (void)n_in; (void)out_size; (void)ws_size;

    const int nquads = (B_DIM + C_DIM) * D_DIM / 4;          // float4s to convert
    cvt_split_kernel<<<nquads / 256, 256, 0, stream>>>(x, cen, xhi, xlo, chi, clo);

    rownorm_kernel<<<(B_DIM + C_DIM) / 8, 256, 0, stream>>>(x, cen, x2, c2);

    dim3 grid(B_DIM / 64, COLB);
    rbf_partial_kernel<<<grid, 256, 0, stream>>>(xhi, xlo, chi, clo, beta, W, x2, c2, partial);

    rbf_final_kernel<<<(B_DIM + 255) / 256, 256, 0, stream>>>(partial, b, out);
}
